// ScatterAndGather_42898133353283
// MI455X (gfx1250) — matrix operand order repaired and checked by reading, not compiled, not run on hardware
//
#include <hip/hip_runtime.h>
#include <hip/hip_bf16.h>
#include <math.h>

#define TT 8
#define NSUBN 50000
#define NNODES 100000
#define DIM 128      // EMBED_DIM
#define CDIM 64      // COMP_DIM
#define D2 256       // 2*DIM
#define C2 128       // 2*CDIM

typedef __attribute__((ext_vector_type(16))) __bf16 v16bf;
typedef __attribute__((ext_vector_type(8)))  __bf16 v8bf;
typedef __attribute__((ext_vector_type(8)))  float  v8f;

// Branchless tanh-form GELU: x * (1 - 1/(exp(2u)+1)), u = 0.79788456(x+0.044715x^3).
// Uses hardware v_exp_f32 / v_rcp_f32; ~8 VALU ops instead of erff's ~40-op
// branchy expansion (which would dominate the whole pipeline's runtime).
__device__ __forceinline__ float gelu_fast(float x) {
  float p = x * x;
  float a = __builtin_fmaf(0.044715f, p, 1.0f);   // 1 + 0.044715 x^2
  float u2 = 1.5957691216057308f * x * a;          // 2 * 0.79788456 * x * a
  float e = __expf(u2);                            // v_exp_f32
  float r = __builtin_amdgcn_rcpf(e + 1.0f);       // v_rcp_f32
  return x - x * r;                                // x * (1 - r)
}

// A-fragment (16x32 bf16, this lane = row lm): per ISA 7.12.2, lane half h
// holds K = {h*8..h*8+7} in elements 0..7 and K = {16+h*8..16+h*8+7} in 8..15.
// Both are 16B-contiguous runs in a row-major tile -> two b128 loads.
__device__ __forceinline__ v16bf load_a_frag(const __bf16* rowp, int k0, int half) {
  const __bf16* p = rowp + k0 + half * 8;
  v8bf lo = *(const v8bf*)(p);
  v8bf hi = *(const v8bf*)(p + 16);
  v16bf a;
#pragma unroll
  for (int i = 0; i < 8; i++) { a[i] = lo[i]; a[i + 8] = hi[i]; }
  return a;
}

// B-fragment (32x16 bf16): lane group h holds K = h*8 + i (i<8) and K = 16 + h*8 + (i-8) (i>=8) for column lm.
// Weights stored transposed (N x K), so this is two runs of 8 contiguous bf16 per lane.
__device__ __forceinline__ v16bf load_b_frag(const __bf16* wt, int K, int ncol,
                                             int k0, int half) {
  const __bf16* p = wt + (size_t)ncol * K + k0 + half * 8;
  v8bf lo = *(const v8bf*)(p);
  v8bf hi = *(const v8bf*)(p + 16);
  v16bf b;
#pragma unroll
  for (int i = 0; i < 8; i++) { b[i] = lo[i]; b[i + 8] = hi[i]; }
  return b;
}

// ---------------- kernel 1: zero the scatter table -------------------------
__global__ void __launch_bounds__(256)
zero_kernel(float4* __restrict__ p, size_t n4) {
  size_t i = (size_t)blockIdx.x * blockDim.x + threadIdx.x;
  if (i < n4) p[i] = make_float4(0.f, 0.f, 0.f, 0.f);
}

// ---------------- kernel 2: per-timestep scatter-add -----------------------
// one lane handles 4 contiguous floats of one subnode row
__global__ void __launch_bounds__(256)
scatter_kernel(const float* __restrict__ x, const int* __restrict__ idx,
               float* __restrict__ S) {
  int gid = blockIdx.x * blockDim.x + threadIdx.x;   // TT*NSUBN*32 threads
  int row = gid >> 5;
  int q   = (gid & 31) * 4;
  int t    = row / NSUBN;
  int node = idx[row];
  float4 v = *(const float4*)(x + (size_t)row * DIM + q);
  float* dst = S + ((size_t)t * NNODES + node) * DIM + q;
  atomicAdd(dst + 0, v.x);
  atomicAdd(dst + 1, v.y);
  atomicAdd(dst + 2, v.z);
  atomicAdd(dst + 3, v.w);
}

// ---------------- kernel 3: weight fp32 (KxN) -> bf16 transposed (NxK) -----
__global__ void __launch_bounds__(256)
convw_kernel(const float* __restrict__ W, unsigned short* __restrict__ WTu,
             int K, int N) {
  int i = blockIdx.x * blockDim.x + threadIdx.x;
  if (i >= K * N) return;
  int k = i / N, n = i % N;
  __bf16* WT = reinterpret_cast<__bf16*>(WTu);
  WT[(size_t)n * K + k] = (__bf16)W[i];
}

// ---------------- kernel 4: LN -> (128->256 GELU) -> (256->64) -------------
// 4 waves/block, each wave owns 16 rows of the 800k virtual rows.
__global__ void __launch_bounds__(128)
mlpd_kernel(const float* __restrict__ S, const float* __restrict__ bd,
            const float* __restrict__ lng, const float* __restrict__ lnb,
            const unsigned short* __restrict__ W1u, const float* __restrict__ b1,
            const unsigned short* __restrict__ W2u, const float* __restrict__ b2,
            float* __restrict__ H) {
  __shared__ __align__(16) __bf16 lnT[4][16][DIM];   // 16 KB
  __shared__ __align__(16) __bf16 h1T[4][16][D2];    // 32 KB
  const __bf16* W1 = reinterpret_cast<const __bf16*>(W1u);
  const __bf16* W2 = reinterpret_cast<const __bf16*>(W2u);
  const int wave = threadIdx.x >> 5;
  const int lane = threadIdx.x & 31;
  const int half = lane >> 4;
  const int lm   = lane & 15;
  const int rowBase = blockIdx.x * 64 + wave * 16;

  // phase 1: LayerNorm(S + bd) -> bf16 tile
  {
    const int d = lane * 4;
    const float g0 = lng[d], g1 = lng[d + 1], g2 = lng[d + 2], g3 = lng[d + 3];
    const float c0 = lnb[d], c1 = lnb[d + 1], c2 = lnb[d + 2], c3 = lnb[d + 3];
    for (int r = 0; r < 16; r++) {
      const int row  = rowBase + r;
      const int node = row % NNODES;
      float4 sv = *(const float4*)(S + (size_t)row * DIM + d);
      float4 bv = *(const float4*)(bd + (size_t)node * DIM + d);
      float v0 = sv.x + bv.x, v1 = sv.y + bv.y, v2 = sv.z + bv.z, v3 = sv.w + bv.w;
      float sum = v0 + v1 + v2 + v3;
      float sq  = v0 * v0 + v1 * v1 + v2 * v2 + v3 * v3;
#pragma unroll
      for (int off = 16; off > 0; off >>= 1) {
        sum += __shfl_xor(sum, off);
        sq  += __shfl_xor(sq, off);
      }
      float mean = sum * (1.0f / DIM);
      float var  = sq * (1.0f / DIM) - mean * mean;
      float rstd = rsqrtf(var + 1e-5f);
      lnT[wave][r][d + 0] = (__bf16)((v0 - mean) * rstd * g0 + c0);
      lnT[wave][r][d + 1] = (__bf16)((v1 - mean) * rstd * g1 + c1);
      lnT[wave][r][d + 2] = (__bf16)((v2 - mean) * rstd * g2 + c2);
      lnT[wave][r][d + 3] = (__bf16)((v3 - mean) * rstd * g3 + c3);
    }
  }
  __syncthreads();

  // phase 2: 16x128 @ 128x256, +bias, GELU -> bf16 tile
  {
    v16bf afr[4];
#pragma unroll
    for (int kt = 0; kt < 4; kt++)
      afr[kt] = load_a_frag(&lnT[wave][lm][0], kt * 32, half);
    for (int nt = 0; nt < 16; nt++) {
      const int n0 = nt * 16;
      const float bias = b1[n0 + lm];
      v8f acc;
#pragma unroll
      for (int j = 0; j < 8; j++) acc[j] = bias;
#pragma unroll
      for (int kt = 0; kt < 4; kt++) {
        v16bf bfr = load_b_frag(W1, DIM, n0 + lm, kt * 32, half);
        acc = __builtin_amdgcn_wmma_f32_16x16x32_bf16(false, afr[kt], false, bfr,
                                                      (short)0, acc, false, false);
      }
#pragma unroll
      for (int j = 0; j < 8; j++)
        h1T[wave][j + half * 8][n0 + lm] = (__bf16)gelu_fast(acc[j]);
    }
  }
  __syncthreads();

  // phase 3: 16x256 @ 256x64, +bias -> H (fp32)
  {
    v16bf afr[8];
#pragma unroll
    for (int kt = 0; kt < 8; kt++)
      afr[kt] = load_a_frag(&h1T[wave][lm][0], kt * 32, half);
    for (int nt = 0; nt < 4; nt++) {
      const int n0 = nt * 16;
      const float bias = b2[n0 + lm];
      v8f acc;
#pragma unroll
      for (int j = 0; j < 8; j++) acc[j] = bias;
#pragma unroll
      for (int kt = 0; kt < 8; kt++) {
        v16bf bfr = load_b_frag(W2, D2, n0 + lm, kt * 32, half);
        acc = __builtin_amdgcn_wmma_f32_16x16x32_bf16(false, afr[kt], false, bfr,
                                                      (short)0, acc, false, false);
      }
#pragma unroll
      for (int j = 0; j < 8; j++)
        H[(size_t)(rowBase + j + half * 8) * CDIM + n0 + lm] = acc[j];
    }
  }
}

// ---------------- kernel 5: gather -> LN -> (64->128 GELU) -> (128->128) ---
__global__ void __launch_bounds__(128)
mlpu_kernel(const float* __restrict__ H, const int* __restrict__ idx,
            const float* __restrict__ lng, const float* __restrict__ lnb,
            const unsigned short* __restrict__ W1u, const float* __restrict__ b1,
            const unsigned short* __restrict__ W2u, const float* __restrict__ b2,
            float* __restrict__ out) {
  __shared__ __align__(16) __bf16 lnT[4][16][CDIM];  // 8 KB
  __shared__ __align__(16) __bf16 u1T[4][16][C2];    // 16 KB
  const __bf16* W1 = reinterpret_cast<const __bf16*>(W1u);
  const __bf16* W2 = reinterpret_cast<const __bf16*>(W2u);
  const int wave = threadIdx.x >> 5;
  const int lane = threadIdx.x & 31;
  const int half = lane >> 4;
  const int lm   = lane & 15;
  const int rowBase = blockIdx.x * 64 + wave * 16;

  // phase 1: gather + LayerNorm -> bf16 tile (64-wide, 2 elems/lane)
  {
    const int c = lane * 2;
    const float g0 = lng[c], g1 = lng[c + 1];
    const float c0 = lnb[c], c1 = lnb[c + 1];
    for (int r = 0; r < 16; r++) {
      const int row  = rowBase + r;
      const int t    = row / NSUBN;
      const int node = idx[row];
      const float* src = H + ((size_t)t * NNODES + node) * CDIM + c;
      float v0 = src[0], v1 = src[1];
      float sum = v0 + v1;
      float sq  = v0 * v0 + v1 * v1;
#pragma unroll
      for (int off = 16; off > 0; off >>= 1) {
        sum += __shfl_xor(sum, off);
        sq  += __shfl_xor(sq, off);
      }
      float mean = sum * (1.0f / CDIM);
      float var  = sq * (1.0f / CDIM) - mean * mean;
      float rstd = rsqrtf(var + 1e-5f);
      lnT[wave][r][c + 0] = (__bf16)((v0 - mean) * rstd * g0 + c0);
      lnT[wave][r][c + 1] = (__bf16)((v1 - mean) * rstd * g1 + c1);
    }
  }
  __syncthreads();

  // phase 2: 16x64 @ 64x128, +bias, GELU -> bf16 tile
  {
    v16bf afr[2];
#pragma unroll
    for (int kt = 0; kt < 2; kt++)
      afr[kt] = load_a_frag(&lnT[wave][lm][0], kt * 32, half);
    for (int nt = 0; nt < 8; nt++) {
      const int n0 = nt * 16;
      const float bias = b1[n0 + lm];
      v8f acc;
#pragma unroll
      for (int j = 0; j < 8; j++) acc[j] = bias;
#pragma unroll
      for (int kt = 0; kt < 2; kt++) {
        v16bf bfr = load_b_frag(W1, CDIM, n0 + lm, kt * 32, half);
        acc = __builtin_amdgcn_wmma_f32_16x16x32_bf16(false, afr[kt], false, bfr,
                                                      (short)0, acc, false, false);
      }
#pragma unroll
      for (int j = 0; j < 8; j++)
        u1T[wave][j + half * 8][n0 + lm] = (__bf16)gelu_fast(acc[j]);
    }
  }
  __syncthreads();

  // phase 3: 16x128 @ 128x128, +bias -> out (fp32)
  {
    v16bf afr[4];
#pragma unroll
    for (int kt = 0; kt < 4; kt++)
      afr[kt] = load_a_frag(&u1T[wave][lm][0], kt * 32, half);
    for (int nt = 0; nt < 8; nt++) {
      const int n0 = nt * 16;
      const float bias = b2[n0 + lm];
      v8f acc;
#pragma unroll
      for (int j = 0; j < 8; j++) acc[j] = bias;
#pragma unroll
      for (int kt = 0; kt < 4; kt++) {
        v16bf bfr = load_b_frag(W2, C2, n0 + lm, kt * 32, half);
        acc = __builtin_amdgcn_wmma_f32_16x16x32_bf16(false, afr[kt], false, bfr,
                                                      (short)0, acc, false, false);
      }
#pragma unroll
      for (int j = 0; j < 8; j++)
        out[(size_t)(rowBase + j + half * 8) * DIM + n0 + lm] = acc[j];
    }
  }
}

// ---------------------------------------------------------------------------
extern "C" void kernel_launch(void* const* d_in, const int* in_sizes, int n_in,
                              void* d_out, int out_size, void* d_ws, size_t ws_size,
                              hipStream_t stream) {
  const float* x       = (const float*)d_in[0];
  const int*   indices = (const int*)d_in[1];
  const float* bd      = (const float*)d_in[2];
  const float* ln_d_g  = (const float*)d_in[3];
  const float* ln_d_b  = (const float*)d_in[4];
  const float* Wd1     = (const float*)d_in[5];
  const float* bd1     = (const float*)d_in[6];
  const float* Wd2     = (const float*)d_in[7];
  const float* bd2     = (const float*)d_in[8];
  const float* ln_u_g  = (const float*)d_in[9];
  const float* ln_u_b  = (const float*)d_in[10];
  const float* Wu1     = (const float*)d_in[11];
  const float* bu1     = (const float*)d_in[12];
  const float* Wu2     = (const float*)d_in[13];
  const float* bu2     = (const float*)d_in[14];
  float* out = (float*)d_out;

  // workspace layout (all offsets 256B-aligned)
  const size_t S_bytes = (size_t)TT * NNODES * DIM * sizeof(float);   // 409.6 MB
  const size_t H_bytes = (size_t)TT * NNODES * CDIM * sizeof(float);  // 204.8 MB
  const size_t W_bytes = (size_t)(128 * 256 + 256 * 64 + 64 * 128 + 128 * 128) * 2;
  if (ws_size < S_bytes + H_bytes + W_bytes) return;  // insufficient scratch

  char* ws = (char*)d_ws;
  float* S = (float*)ws;
  float* H = (float*)(ws + S_bytes);
  unsigned short* WTd1 = (unsigned short*)(ws + S_bytes + H_bytes);
  unsigned short* WTd2 = WTd1 + 128 * 256;
  unsigned short* WTu1 = WTd2 + 256 * 64;
  unsigned short* WTu2 = WTu1 + 64 * 128;

  // 1) zero scatter table
  {
    size_t n4 = (size_t)TT * NNODES * DIM / 4;  // 25.6M float4
    int blocks = (int)((n4 + 255) / 256);
    zero_kernel<<<blocks, 256, 0, stream>>>((float4*)S, n4);
  }
  // 2) weight conversion (transposed bf16)
  convw_kernel<<<(128 * 256 + 255) / 256, 256, 0, stream>>>(Wd1, WTd1, 128, 256);
  convw_kernel<<<(256 * 64 + 255) / 256, 256, 0, stream>>>(Wd2, WTd2, 256, 64);
  convw_kernel<<<(64 * 128 + 255) / 256, 256, 0, stream>>>(Wu1, WTu1, 64, 128);
  convw_kernel<<<(128 * 128 + 255) / 256, 256, 0, stream>>>(Wu2, WTu2, 128, 128);
  // 3) scatter-add
  {
    int threads = TT * NSUBN * 32;  // 12.8M
    scatter_kernel<<<threads / 256, 256, 0, stream>>>(x, indices, S);
  }
  // 4) LN + mlp_d (800k rows, 64 rows/block)
  mlpd_kernel<<<(TT * NNODES) / 64, 128, 0, stream>>>(S, bd, ln_d_g, ln_d_b,
                                                      WTd1, bd1, WTd2, bd2, H);
  // 5) gather + LN + mlp_u (400k rows, 64 rows/block)
  mlpu_kernel<<<(TT * NSUBN) / 64, 128, 0, stream>>>(H, indices, ln_u_g, ln_u_b,
                                                     WTu1, bu1, WTu2, bu2, out);
}
